// LightweightMSADCF_73203422593132
// MI455X (gfx1250) — matrix operand order repaired and checked by reading, not compiled, not run on hardware
//
#include <hip/hip_runtime.h>

typedef __attribute__((ext_vector_type(16))) __bf16 v16bf;
typedef __attribute__((ext_vector_type(8)))  __bf16 bf16x8;
typedef __attribute__((ext_vector_type(8)))  float  v8f;
typedef __attribute__((ext_vector_type(4)))  unsigned int v4u;
typedef __attribute__((ext_vector_type(8)))  int v8i;
typedef __attribute__((ext_vector_type(4)))  int v4i;

union Frag {
  v16bf  v;
  bf16x8 h[2];
};

__device__ __forceinline__ v8f wmma_bf16(v16bf a, v16bf b, v8f c) {
  // D = A(16x32 bf16) * B(32x16 bf16) + C(16x16 f32)
  return __builtin_amdgcn_wmma_f32_16x16x32_bf16(false, a, false, b, (short)0, c,
                                                 false, false);
}

// ---------------------------------------------------------------------------
// TDM 1-D copy: global -> LDS, nelem4 dwords (<= 65535), one descriptor.
// D# layout per CDNA5 ISA:
//  group0: [1:0]=count=1, [63:32]=lds_addr, [120:64]=global_addr, [127:126]=2
//  group1: data_size=2 (4B), tensor_dim0=nelem4, tensor_dim1=1,
//          tile_dim0=nelem4, tile_dim1=1, dim0_stride=nelem4
// Tracked by TENSORcnt; EXEC ignored (issue from one wave only).
// amdgpu-toolchain (clang-23) 6-arg builtin form.
// ---------------------------------------------------------------------------
__device__ __forceinline__ void tdm_copy_1d(unsigned int lds_addr,
                                            const void* gsrc,
                                            unsigned int nelem4) {
  unsigned long long ga = (unsigned long long)(size_t)gsrc;
  v4u g0;
  g0.x = 1u;                                             // count=1
  g0.y = lds_addr;                                       // lds_addr
  g0.z = (unsigned int)ga;                               // global_addr[31:0]
  g0.w = (unsigned int)((ga >> 32) & 0x01FFFFFFu) | (2u << 30);  // ga[56:32], type=2
  v8i g1;
  g1[0] = (int)(2u << 16);                               // mask=0, data_size=4B
  g1[1] = (int)((nelem4 & 0xFFFFu) << 16);               // tensor_dim0[15:0]
  g1[2] = (int)(((nelem4 >> 16) & 0xFFFFu) | (1u << 16)); // td0[31:16], tensor_dim1=1
  g1[3] = (int)((nelem4 & 0xFFFFu) << 16);               // tile_dim0
  g1[4] = 1;                                             // tile_dim1=1, tile_dim2=0
  g1[5] = (int)nelem4;                                   // dim0_stride[31:0]
  g1[6] = 0;                                             // stride hi / dim1_stride
  g1[7] = 0;
  v4i z4 = {0, 0, 0, 0};
  v8i z8 = {0, 0, 0, 0, 0, 0, 0, 0};
  __builtin_amdgcn_tensor_load_to_lds(g0, g1, z4, z4, z8, 0);
}

#define HWIMG 16384  // 128*128
#define IMGW  128

// ===========================================================================
// Kernel 1: global average pool of concat([rgb, depth]) -> pooled[64][128]
// ===========================================================================
__global__ void __launch_bounds__(256) pool_kernel(const float* __restrict__ rgb,
                                                   const float* __restrict__ depth,
                                                   float* __restrict__ pooled) {
  int q  = blockIdx.x;            // 0..8191 : b*128 + ch2
  int b  = q >> 7;
  int c2 = q & 127;
  const float* src = (c2 < 64) ? (rgb   + (size_t)(b * 64 + c2)      * HWIMG)
                               : (depth + (size_t)(b * 64 + (c2-64)) * HWIMG);
  float s = 0.f;
  for (int i = threadIdx.x; i < HWIMG; i += 256) s += src[i];
  __shared__ float red[256];
  red[threadIdx.x] = s;
  __syncthreads();
  for (int off = 128; off > 0; off >>= 1) {
    if (threadIdx.x < off) red[threadIdx.x] += red[threadIdx.x + off];
    __syncthreads();
  }
  if (threadIdx.x == 0) pooled[q] = red[0] * (1.0f / 16384.0f);
}

// ===========================================================================
// Kernel 2: per-image MLP -> dynamic kernels -> softmax attn -> weighted
// kernel -> pack into WMMA A-fragment layout (bf16). One block per b.
// ===========================================================================
__global__ void __launch_bounds__(256) kgen_kernel(const float* __restrict__ pooled,
                                                   const float* __restrict__ w1,
                                                   const float* __restrict__ b1,
                                                   const float* __restrict__ w2,
                                                   const float* __restrict__ b2,
                                                   __bf16* __restrict__ wk_pack) {
  int b = blockIdx.x;
  int t = threadIdx.x;
  __shared__ float pl[128];
  __shared__ float hid[64];
  __shared__ float nsq[4];
  __shared__ float wkbuf[576];

  if (t < 128) pl[t] = pooled[b * 128 + t];
  if (t < 4)   nsq[t] = 0.f;
  for (int i = t; i < 576; i += 256) wkbuf[i] = 0.f;
  __syncthreads();

  if (t < 64) {
    float a = b1[t];
    for (int j = 0; j < 128; ++j) a += w1[t * 128 + j] * pl[j];
    hid[t] = a > 0.f ? a : 0.f;
  }
  __syncthreads();

  float val[9];
#pragma unroll
  for (int j = 0; j < 9; ++j) {
    int r = j * 256 + t;          // 0..2303
    float a = b2[r];
    for (int q = 0; q < 64; ++q) a += w2[r * 64 + q] * hid[q];
    val[j] = a;
    atomicAdd(&nsq[r / 576], a * a);   // ds_add_f32
  }
  __syncthreads();

  // softmax over k (redundant per thread, tiny)
  float norm[4], attn[4];
  float mx = -1e30f;
  for (int k = 0; k < 4; ++k) { norm[k] = sqrtf(nsq[k]); mx = fmaxf(mx, norm[k]); }
  float den = 0.f;
  for (int k = 0; k < 4; ++k) { attn[k] = expf(norm[k] - mx); den += attn[k]; }
  float inv = 1.0f / den;
  for (int k = 0; k < 4; ++k) attn[k] *= inv;

#pragma unroll
  for (int j = 0; j < 9; ++j) {
    int r = j * 256 + t;
    atomicAdd(&wkbuf[r % 576], attn[r / 576] * val[j]);  // ds_add_f32
  }
  __syncthreads();

  // pack wk[b][ci][tap] : output channel M == b
  // A-frag mapping: kl=k&31 -> laneHi=(kl>>3)&1, j=(kl&7)+((kl>>4)<<3)
  int Mt = b >> 4, Ml = b & 15;
  for (int idx = t; idx < 576; idx += 256) {
    int ci = idx / 9, tap = idx % 9;
    int k  = tap * 64 + ci;
    int s  = k >> 5, kl = k & 31;
    int laneHi = (kl >> 3) & 1;
    int j    = (kl & 7) + ((kl >> 4) << 3);
    int lane = Ml + (laneHi << 4);
    wk_pack[((Mt * 18 + s) * 32 + lane) * 16 + j] = (__bf16)wkbuf[idx];
  }
}

// ===========================================================================
// Kernel 3: pack static conv weights into A-fragment layout + fold BN affine
// One block per output channel co.
// ===========================================================================
__global__ void __launch_bounds__(256) pack_static_kernel(
    const float* __restrict__ fd_w, const float* __restrict__ fu_w,
    const float* __restrict__ fd_b, const float* __restrict__ fd_scale,
    const float* __restrict__ fd_bias, const float* __restrict__ fd_mean,
    const float* __restrict__ fd_var,
    const float* __restrict__ fu_b, const float* __restrict__ fu_scale,
    const float* __restrict__ fu_bias, const float* __restrict__ fu_mean,
    const float* __restrict__ fu_var,
    __bf16* __restrict__ fd_pack, __bf16* __restrict__ fu_pack,
    float* __restrict__ fdA, float* __restrict__ fdD,
    float* __restrict__ fuA, float* __restrict__ fuD) {
  int co = blockIdx.x;
  int t  = threadIdx.x;
  int Mt = co >> 4, Ml = co & 15;

  for (int idx = t; idx < 576; idx += 256) {
    int k  = (idx % 9) * 64 + (idx / 9);   // K = tap*64 + ci
    int s  = k >> 5, kl = k & 31;
    int laneHi = (kl >> 3) & 1;
    int j    = (kl & 7) + ((kl >> 4) << 3);
    int lane = Ml + (laneHi << 4);
    fd_pack[((Mt * 18 + s) * 32 + lane) * 16 + j] = (__bf16)fd_w[co * 576 + idx];
  }
  if (t < 128) {
    int k = t;                              // K = input channel of 1x1
    int s = k >> 5, kl = k & 31;
    int laneHi = (kl >> 3) & 1;
    int j    = (kl & 7) + ((kl >> 4) << 3);
    int lane = Ml + (laneHi << 4);
    fu_pack[((Mt * 4 + s) * 32 + lane) * 16 + j] = (__bf16)fu_w[co * 128 + k];
  }
  if (co == 0 && t < 64) {
    float ia = fd_scale[t] * rsqrtf(fd_var[t] + 1e-5f);
    fdA[t] = ia;
    fdD[t] = fd_bias[t] + (fd_b[t] - fd_mean[t]) * ia;
    float ib = fu_scale[t] * rsqrtf(fu_var[t] + 1e-5f);
    fuA[t] = ib;
    fuD[t] = fu_bias[t] + (fu_b[t] - fu_mean[t]) * ib;
  }
}

// ===========================================================================
// Kernel 4: fully fused conv pipeline. One block = one batch image x one
// 16x16 pixel tile. 8 waves, WMMA bf16 implicit GEMM with TWO independent
// accumulator chains per wave (same M-tile, adjacent pixel rows) so the
// XDL pipeline alternates independent WMMAs instead of stalling, and each
// A fragment is reused for two WMMAs. Weight packs are DMA'd into LDS by
// the Tensor Data Mover (TENSORcnt) overlapped with the cooperative
// f32->bf16 tile load.
//   Phase A: freq = relu(bn(conv3x3(depth))) -> stage[px][64..127]
//   Phase B: dyn  = conv3x3(rgb, wk)         -> stage[px][ 0.. 63]
//   Phase C: out  = relu(bn(fu_w @ stage))   -> global f32
// LDS: 41472 (tile) + 65536 (stage) + 73728 (weights) + 1024 = ~182 KB
// -> one workgroup per 320KB WGP (2 waves/SIMD; ILP-oriented schedule).
// ===========================================================================
__global__ void __launch_bounds__(256) fused_conv_kernel(
    const float* __restrict__ rgb, const float* __restrict__ depth,
    const __bf16* __restrict__ wk_pack, const __bf16* __restrict__ fd_pack,
    const __bf16* __restrict__ fu_pack,
    const float* __restrict__ fdAp, const float* __restrict__ fdDp,
    const float* __restrict__ fuAp, const float* __restrict__ fuDp,
    float* __restrict__ out) {
  // 18x18 spatial halo tile, channel-minor (ci contiguous -> b128 B-frags)
  __shared__ __attribute__((aligned(16))) __bf16 lds_in[18 * 18 * 64];   // 41472 B
  // staged conv outputs, channel-minor: [pixel 0..255][ch 0..127]
  __shared__ __attribute__((aligned(16))) __bf16 stage[256 * 128];       // 65536 B
  // TDM-staged A-fragment weight pack for the current phase
  __shared__ __attribute__((aligned(16))) __bf16 lds_w[4 * 18 * 32 * 16]; // 73728 B
  __shared__ float fdA[64], fdD[64], fuA[64], fuD[64];

  const int tid  = threadIdx.x;
  const int lane = tid & 31;
  const int wave = tid >> 5;
  const int blk  = blockIdx.x;            // 0..4095
  const int b    = blk >> 6;
  const int tile = blk & 63;
  const int y0   = (tile >> 3) << 4;
  const int x0   = (tile & 7) << 4;

  const float* rgb_b   = rgb   + (size_t)b * 64 * HWIMG;
  const float* depth_b = depth + (size_t)b * 64 * HWIMG;
  const unsigned int lds_w_addr = (unsigned int)(size_t)(&lds_w[0]);

  if (tid < 64) { fdA[tid] = fdAp[tid]; fdD[tid] = fdDp[tid];
                  fuA[tid] = fuAp[tid]; fuD[tid] = fuDp[tid]; }

  // cooperative f32->bf16 tile load with zero-pad halo
  auto load_tile = [&](const float* src) {
    for (int task = tid; task < 64 * 18; task += 256) {
      int ch  = task / 18;
      int row = task % 18;
      int gy  = y0 - 1 + row;
      bool rowok = (gy >= 0) && (gy < 128);
      const float* gp = src + ch * HWIMG + gy * IMGW;
      __bf16* dp = lds_in + row * (18 * 64) + ch;
#pragma unroll
      for (int col = 0; col < 18; ++col) {
        int gx = x0 - 1 + col;
        float v = (rowok && gx >= 0 && gx < 128) ? gp[gx] : 0.f;
        dp[col * 64] = (__bf16)v;
      }
    }
  };

  // 3x3 conv over the LDS tile; A fragments from TDM-staged lds_w.
  // Each wave: 4 iterations of (Mtile m, pixel rows t0, t0+1).
  auto conv_phase = [&](int co_off, bool act) {
    for (int p = wave; p < 32; p += 8) {
      int m  = p >> 3;            // output-channel tile 0..3
      int t0 = (p & 7) * 2;       // even pixel row; chain 1 does t0+1
      v8f acc0 = {};
      v8f acc1 = {};
      for (int s = 0; s < 18; ++s) {            // K = 576 = 18 * 32
        int tap = s >> 1;
        int dy = tap / 3, dx = tap % 3;
        Frag A;
        const __bf16* ap = lds_w + ((m * 18 + s) * 32 + lane) * 16;
        A.h[0] = *(const bf16x8*)(ap);
        A.h[1] = *(const bf16x8*)(ap + 8);
        // B fragments: lane<16 -> K 0..7,16..23, lane>=16 -> K 8..15,24..31
        int px = (lane & 15) + dx;
        int cb = ((s & 1) << 5) + ((lane >> 4) << 3);
        const __bf16* bp0 = lds_in + ((t0 + dy) * 18 + px) * 64 + cb;
        Frag B0, B1;
        B0.h[0] = *(const bf16x8*)(bp0);
        B0.h[1] = *(const bf16x8*)(bp0 + 16);
        B1.h[0] = *(const bf16x8*)(bp0 + 18 * 64);     // row t0+1
        B1.h[1] = *(const bf16x8*)(bp0 + 18 * 64 + 16);
        acc0 = wmma_bf16(A.v, B0.v, acc0);
        acc1 = wmma_bf16(A.v, B1.v, acc1);
      }
      int nl = lane & 15, hl = lane >> 4;
#pragma unroll
      for (int r = 0; r < 8; ++r) {             // C/D: M = r + 8*hl
        int ch = m * 16 + r + hl * 8;
        float v0 = acc0[r];
        float v1 = acc1[r];
        if (act) {
          v0 = v0 * fdA[ch] + fdD[ch]; v0 = v0 > 0.f ? v0 : 0.f;
          v1 = v1 * fdA[ch] + fdD[ch]; v1 = v1 > 0.f ? v1 : 0.f;
        }
        stage[(t0 * 16 + nl) * 128 + co_off + ch]       = (__bf16)v0;
        stage[((t0 + 1) * 16 + nl) * 128 + co_off + ch] = (__bf16)v1;
      }
    }
  };

  // ---- Phase A: depth -> freq features; TDM fd_pack in parallel ----
  if (wave == 0) tdm_copy_1d(lds_w_addr, fd_pack, 73728u / 4u);
  load_tile(depth_b);
  // prefetch rgb tile into cache while depth phase runs
  for (int task = tid; task < 64 * 18; task += 256) {
    int ch = task / 18, row = task % 18;
    int gy = y0 - 1 + row;
    if (gy >= 0 && gy < 128)
      __builtin_prefetch(rgb_b + ch * HWIMG + gy * IMGW + x0, 0, 1);
  }
  if (wave == 0) __builtin_amdgcn_s_wait_tensorcnt(0);
  __syncthreads();
  conv_phase(/*co_off=*/64, /*act=*/true);
  __syncthreads();

  // ---- Phase B: rgb -> dynamic features; TDM wk_pack in parallel ----
  if (wave == 0) tdm_copy_1d(lds_w_addr, wk_pack, 73728u / 4u);
  load_tile(rgb_b);
  if (wave == 0) __builtin_amdgcn_s_wait_tensorcnt(0);
  __syncthreads();
  conv_phase(/*co_off=*/0, /*act=*/false);
  __syncthreads();

  // ---- Phase C: 1x1 fusion conv over stage, K = 128 ----
  if (wave == 0) {
    tdm_copy_1d(lds_w_addr, fu_pack, 16384u / 4u);
    __builtin_amdgcn_s_wait_tensorcnt(0);
  }
  __syncthreads();
  for (int p = wave; p < 32; p += 8) {
    int m  = p >> 3;
    int t0 = (p & 7) * 2;
    v8f acc0 = {};
    v8f acc1 = {};
#pragma unroll
    for (int s = 0; s < 4; ++s) {
      Frag A;
      const __bf16* ap = lds_w + ((m * 4 + s) * 32 + lane) * 16;
      A.h[0] = *(const bf16x8*)(ap);
      A.h[1] = *(const bf16x8*)(ap + 8);
      int nl = lane & 15;
      int cb = (s << 5) + ((lane >> 4) << 3);
      const __bf16* bp0 = stage + (t0 * 16 + nl) * 128 + cb;
      Frag B0, B1;
      B0.h[0] = *(const bf16x8*)(bp0);
      B0.h[1] = *(const bf16x8*)(bp0 + 16);
      B1.h[0] = *(const bf16x8*)(bp0 + 16 * 128);      // row t0+1
      B1.h[1] = *(const bf16x8*)(bp0 + 16 * 128 + 16);
      acc0 = wmma_bf16(A.v, B0.v, acc0);
      acc1 = wmma_bf16(A.v, B1.v, acc1);
    }
    int nl = lane & 15, hl = lane >> 4;
    int x = x0 + nl;
#pragma unroll
    for (int r = 0; r < 8; ++r) {
      int co = m * 16 + r + hl * 8;
      float v0 = acc0[r] * fuA[co] + fuD[co];
      float v1 = acc1[r] * fuA[co] + fuD[co];
      v0 = v0 > 0.f ? v0 : 0.f;
      v1 = v1 > 0.f ? v1 : 0.f;
      size_t base = (size_t)((b * 64 + co) * 128) * 128 + x;
      out[base + (size_t)(y0 + t0) * 128]     = v0;
      out[base + (size_t)(y0 + t0 + 1) * 128] = v1;
    }
  }
}

// ===========================================================================
extern "C" void kernel_launch(void* const* d_in, const int* in_sizes, int n_in,
                              void* d_out, int out_size, void* d_ws, size_t ws_size,
                              hipStream_t stream) {
  const float* rgb      = (const float*)d_in[0];
  const float* depth    = (const float*)d_in[1];
  const float* w1       = (const float*)d_in[2];
  const float* b1       = (const float*)d_in[3];
  const float* w2       = (const float*)d_in[4];
  const float* b2       = (const float*)d_in[5];
  const float* fd_w     = (const float*)d_in[6];
  const float* fd_b     = (const float*)d_in[7];
  const float* fd_scale = (const float*)d_in[8];
  const float* fd_bias  = (const float*)d_in[9];
  const float* fd_mean  = (const float*)d_in[10];
  const float* fd_var   = (const float*)d_in[11];
  const float* fu_w     = (const float*)d_in[12];
  const float* fu_b     = (const float*)d_in[13];
  const float* fu_scale = (const float*)d_in[14];
  const float* fu_bias  = (const float*)d_in[15];
  const float* fu_mean  = (const float*)d_in[16];
  const float* fu_var   = (const float*)d_in[17];
  float* out = (float*)d_out;

  char* ws = (char*)d_ws;
  float*  pooled  = (float*)(ws);                       //  32768 B
  __bf16* wk_pack = (__bf16*)(ws + 32768);              //  73728 B
  __bf16* fd_pack = (__bf16*)(ws + 32768 + 73728);      //  73728 B
  __bf16* fu_pack = (__bf16*)(ws + 32768 + 2 * 73728);  //  16384 B
  float*  fdA     = (float*)(ws + 32768 + 2 * 73728 + 16384);
  float*  fdD     = fdA + 64;
  float*  fuA     = fdA + 128;
  float*  fuD     = fdA + 192;

  pool_kernel<<<8192, 256, 0, stream>>>(rgb, depth, pooled);
  kgen_kernel<<<64, 256, 0, stream>>>(pooled, w1, b1, w2, b2, wk_pack);
  pack_static_kernel<<<64, 256, 0, stream>>>(
      fd_w, fu_w, fd_b, fd_scale, fd_bias, fd_mean, fd_var,
      fu_b, fu_scale, fu_bias, fu_mean, fu_var,
      fd_pack, fu_pack, fdA, fdD, fuA, fuD);
  fused_conv_kernel<<<4096, 256, 0, stream>>>(
      rgb, depth, wk_pack, fd_pack, fu_pack, fdA, fdD, fuA, fuD, out);
}
